// Router_32384053412032
// MI455X (gfx1250) — matrix operand order repaired and checked by reading, not compiled, not run on hardware
//
#include <hip/hip_runtime.h>

// Router: gates = x[16384,2048] @ W[2048,64] + b[64]; top-2 + softmax(2).
// bf16x3 split-precision WMMA GEMM; x streamed via TDM (tensor_load_to_lds)
// with a 4-stage per-wave pipeline to keep ~12KB per wave in flight.

typedef __attribute__((ext_vector_type(16))) __bf16       v16bf;
typedef __attribute__((ext_vector_type(8)))  float        v8f;
typedef __attribute__((ext_vector_type(8)))  unsigned int v8u;
typedef __attribute__((ext_vector_type(4)))  unsigned int u32x4;
typedef __attribute__((ext_vector_type(8)))  int          i32x8;
typedef __attribute__((ext_vector_type(4)))  int          i32x4;

union FragU {
    v8u   u;
    v16bf b;
};

static constexpr int D      = 2048;      // hidden dim (K)
static constexpr int E      = 64;        // experts (N)
static constexpr int ROWS   = 16384;     // B*S (M)
static constexpr int KSTEP  = 32;        // bf16 WMMA K
static constexpr int KTILES = D / KSTEP; // 64
static constexpr int NTILES = E / 16;    // 4

static constexpr int KCHUNK   = 64;            // K per TDM chunk (2 WMMA steps)
static constexpr int NCHUNKS  = D / KCHUNK;    // 32
static constexpr int STAGES   = 4;             // LDS pipeline depth
static constexpr int ROW_DW   = KCHUNK + 10;   // 74 dwords: +10 pad (TDM pad)
static constexpr int STAGE_DW = 16 * ROW_DW;   // 1184 dwords / stage
static constexpr int WPB      = 2;             // waves per block

// ---------------------------------------------------------------------------
// Prologue: repack W (f32 [D,E]) into bf16 hi/lo B-fragment planes.
// tile t = kt*4+nt; lane l (n=l&15 column, h=l>>4 K-half), elem e:
//   W[kt*32 + h*16 + e][nt*16 + n]; 8 dwords per lane, contiguous.
// ---------------------------------------------------------------------------
__global__ __launch_bounds__(256) void router_prep_w(
    const float* __restrict__ W,
    unsigned int* __restrict__ hiBuf,
    unsigned int* __restrict__ loBuf)
{
    const int tid = blockIdx.x * 256 + threadIdx.x;
    const int t   = tid >> 5;
    const int l   = tid & 31;
    if (t >= KTILES * NTILES) return;

    const int kt  = t >> 2;
    const int nt  = t & 3;
    const int h   = l >> 4;
    const int n   = l & 15;
    const int col = nt * 16 + n;

    FragU hi, lo;
#pragma unroll
    for (int e = 0; e < 16; ++e) {
        const int   k = kt * KSTEP + h * 16 + e;
        const float f = W[(size_t)k * E + col];
        const __bf16 hb = (__bf16)f;
        const float  r  = f - (float)hb;
        hi.b[e] = hb;
        lo.b[e] = (__bf16)r;
    }
    *(v8u*)(hiBuf + ((size_t)(t * 32 + l)) * 8) = hi.u;
    *(v8u*)(loBuf + ((size_t)(t * 32 + l)) * 8) = lo.u;
}

// ---------------------------------------------------------------------------
// Main kernel: each wave = 16 rows x 64 experts, K pipelined through LDS
// by the Tensor Data Mover.
// ---------------------------------------------------------------------------
__global__ __launch_bounds__(WPB * 32) void router_main(
    const float* __restrict__ x,
    const float* __restrict__ bias,
    const unsigned int* __restrict__ hiBuf,
    const unsigned int* __restrict__ loBuf,
    float* __restrict__ wOut,
    int*   __restrict__ idxOut)
{
    __shared__ __align__(16) float smem[WPB * STAGES * STAGE_DW]; // 37.9 KB

    const int l    = threadIdx.x & 31;
    const int w    = threadIdx.x >> 5;             // wave in block
    const int wave = blockIdx.x * WPB + w;         // 0..1023
    const int h    = l >> 4;
    const int n    = l & 15;
    const int rowBase = wave * 16;

    const float* gbase = x + (size_t)rowBase * D;

    // ---- TDM issue: one 16 x 64 f32 tile -> LDS stage, padded 10 dw / row.
    auto tdm_issue = [&](int chunk, int stage) {
        const float* gptr = gbase + (size_t)chunk * KCHUNK;
        unsigned lo32 = (unsigned)(uintptr_t)gptr;
        unsigned hi32 = (unsigned)((uintptr_t)gptr >> 32);
        unsigned lds  = (unsigned)(uintptr_t)(smem + (w * STAGES + stage) * STAGE_DW);
        lo32 = __builtin_amdgcn_readfirstlane(lo32);
        hi32 = __builtin_amdgcn_readfirstlane(hi32);
        lds  = __builtin_amdgcn_readfirstlane(lds);

        // D# group0: count=1 | lds_addr | global_addr | type=2
        u32x4 g0 = { 1u, lds, lo32, (hi32 & 0x01FFFFFFu) | 0x80000000u };
        // D# group1:
        //  d0: data_size=4B(2<<16) | pad_en(1<<20) | interval=64dw(5<<22)
        //      | pad=10dw(9<<25)
        //  d1: tensor_dim0[15:0]<<16 (2048)
        //  d2: tensor_dim1[15:0]<<16 (16384)   d3: tile_dim0<<16 (64)
        //  d4: tile_dim1 (16)                  d5: tensor_dim0_stride (2048)
        i32x8 g1 = { 0x13520000, (int)0x08000000u, (int)0x40000000u,
                     0x00400000, 16, 2048, 0, 0 };
        i32x4 z4 = { 0, 0, 0, 0 };
        i32x8 z8 = { 0, 0, 0, 0, 0, 0, 0, 0 };
        __builtin_amdgcn_tensor_load_to_lds(g0, g1, z4, z4, z8, 0);
    };

    v8f acc[NTILES] = {};

    // ---- compute one chunk (2 WMMA K-steps) out of an LDS stage.
    auto compute_chunk = [&](int c, int stage) {
        const float* sA = smem + (w * STAGES + stage) * STAGE_DW
                               + n * ROW_DW + h * 16;
#pragma unroll
        for (int kk = 0; kk < 2; ++kk) {
            FragU ahi, alo;
#pragma unroll
            for (int v = 0; v < 8; ++v) {
                const int off = kk * 32 + v * 2;
                const float2 f = *(const float2*)(sA + off);   // ds_load_b64
                const __bf16 h0 = (__bf16)f.x;
                const __bf16 h1 = (__bf16)f.y;
                const float  r0 = f.x - (float)h0;
                const float  r1 = f.y - (float)h1;
                ahi.b[2 * v]     = h0;
                ahi.b[2 * v + 1] = h1;
                alo.b[2 * v]     = (__bf16)r0;
                alo.b[2 * v + 1] = (__bf16)r1;
            }
            const int itg = c * 2 + kk;
#pragma unroll
            for (int nt = 0; nt < NTILES; ++nt) {
                const int t = itg * NTILES + nt;
                FragU bhi, blo;
                bhi.u = *(const v8u*)(hiBuf + ((size_t)(t * 32 + l)) * 8);
                blo.u = *(const v8u*)(loBuf + ((size_t)(t * 32 + l)) * 8);
                acc[nt] = __builtin_amdgcn_wmma_f32_16x16x32_bf16(
                    false, ahi.b, false, bhi.b, (short)0, acc[nt], false, false);
                acc[nt] = __builtin_amdgcn_wmma_f32_16x16x32_bf16(
                    false, ahi.b, false, blo.b, (short)0, acc[nt], false, false);
                acc[nt] = __builtin_amdgcn_wmma_f32_16x16x32_bf16(
                    false, alo.b, false, bhi.b, (short)0, acc[nt], false, false);
            }
        }
    };

    // ---- software pipeline: depth STAGES-1 of TDM prefetch in flight.
    tdm_issue(0, 0);
    tdm_issue(1, 1);
    tdm_issue(2, 2);
    for (int c = 0; c < NCHUNKS - (STAGES - 1); ++c) {
        asm volatile("s_wait_dscnt 0x0" ::: "memory"); // stage reads drained
        tdm_issue(c + STAGES - 1, (c + STAGES - 1) % STAGES);
        __builtin_amdgcn_s_wait_tensorcnt(STAGES - 1); // chunk c landed
        asm volatile("" ::: "memory");
        compute_chunk(c, c % STAGES);
    }
    __builtin_amdgcn_s_wait_tensorcnt(2);
    asm volatile("" ::: "memory");
    compute_chunk(NCHUNKS - 3, (NCHUNKS - 3) % STAGES);
    __builtin_amdgcn_s_wait_tensorcnt(1);
    asm volatile("" ::: "memory");
    compute_chunk(NCHUNKS - 2, (NCHUNKS - 2) % STAGES);
    __builtin_amdgcn_s_wait_tensorcnt(0);
    asm volatile("" ::: "memory");
    compute_chunk(NCHUNKS - 1, (NCHUNKS - 1) % STAGES);

    // ---- epilogue: +bias, top-2 over 64 experts per row, softmax over 2.
    float bv[NTILES];
#pragma unroll
    for (int nt = 0; nt < NTILES; ++nt) bv[nt] = bias[nt * 16 + n];

#pragma unroll
    for (int r = 0; r < 8; ++r) {
        float v0 = -__builtin_inff(); int i0 = 0x7fffffff;
        float v1 = -__builtin_inff(); int i1 = 0x7fffffff;
#pragma unroll
        for (int nt = 0; nt < NTILES; ++nt) {
            const float val = acc[nt][r] + bv[nt];
            const int   idx = nt * 16 + n;
            if (val > v0 || (val == v0 && idx < i0)) {
                v1 = v0; i1 = i0; v0 = val; i0 = idx;
            } else if (val > v1 || (val == v1 && idx < i1)) {
                v1 = val; i1 = idx;
            }
        }
#pragma unroll
        for (int s = 1; s <= 8; s <<= 1) {
            const float ov0 = __shfl_xor(v0, s, 32);
            const int   oi0 = __shfl_xor(i0, s, 32);
            const float ov1 = __shfl_xor(v1, s, 32);
            const int   oi1 = __shfl_xor(i1, s, 32);
            float t0, t1; int j0, j1;
            const bool aFirst = (v0 > ov0) || (v0 == ov0 && i0 < oi0);
            if (aFirst) {
                t0 = v0; j0 = i0;
                const bool sec = (ov0 > v1) || (ov0 == v1 && oi0 < i1);
                t1 = sec ? ov0 : v1; j1 = sec ? oi0 : i1;
            } else {
                t0 = ov0; j0 = oi0;
                const bool sec = (v0 > ov1) || (v0 == ov1 && i0 < oi1);
                t1 = sec ? v0 : ov1; j1 = sec ? i0 : oi1;
            }
            v0 = t0; i0 = j0; v1 = t1; i1 = j1;
        }
        if (n == 0) {
            const int row = rowBase + r + h * 8;
            const float e  = __expf(v1 - v0);    // v1 <= v0
            const float w0 = 1.0f / (1.0f + e);
            wOut[row * 2 + 0]   = w0;
            wOut[row * 2 + 1]   = 1.0f - w0;
            idxOut[row * 2 + 0] = i0;
            idxOut[row * 2 + 1] = i1;
        }
    }
}

extern "C" void kernel_launch(void* const* d_in, const int* in_sizes, int n_in,
                              void* d_out, int out_size, void* d_ws, size_t ws_size,
                              hipStream_t stream)
{
    const float* x = (const float*)d_in[0];   // [4,4096,2048]
    const float* W = (const float*)d_in[1];   // [2048,64]
    const float* b = (const float*)d_in[2];   // [64]

    float* wOut   = (float*)d_out;            // 32768 f32 weights
    int*   idxOut = (int*)d_out + ROWS * 2;   // 32768 i32 indices

    unsigned int* hiBuf = (unsigned int*)d_ws;                // 256 KB
    unsigned int* loBuf = hiBuf + KTILES * NTILES * 32 * 8;   // +256 KB

    router_prep_w<<<32, 256, 0, stream>>>(W, hiBuf, loBuf);
    router_main<<<ROWS / 16 / WPB, WPB * 32, 0, stream>>>(
        x, b, hiBuf, loBuf, wOut, idxOut);
}
